// LSTMEncoder_44358422233670
// MI455X (gfx1250) — matrix operand order repaired and checked by reading, not compiled, not run on hardware
//
#include <hip/hip_runtime.h>
#include <hip/hip_bf16.h>

// ---------------------------------------------------------------------------
// LSTM encoder for MI455X (gfx1250, wave32, WMMA).
//   1) cvt kernels: x, W_*, R_* fp32 -> bf16 (vectorized float4 -> 4x bf16).
//   2) proj_kernel: 4 GEMMs [B*T,I]x[I,H], v_wmma_f32_16x16x32_bf16,
//      32x64 tile per wave, register double-buffered K pipeline,
//      bias folded, stored bf16 as proj[g][t][b][h].
//   3) step_kernel x T: h_prev @ R_g^T for all 4 gates per wave (shared A),
//      double-buffered K pipeline, global_prefetch of epilogue operands,
//      fused LSTM cell update + running mean.
//   4) final_kernel: h_bar = hacc / T.
// ---------------------------------------------------------------------------

typedef __attribute__((ext_vector_type(16))) __bf16 bf16x16;
typedef __attribute__((ext_vector_type(8)))  float  floatx8;

union FragU {
    bf16x16 v;
    uint4   q[2];
    unsigned short u[16];
};

__device__ __forceinline__ unsigned short f2bf(float f) {
    unsigned u = __float_as_uint(f);
    unsigned r = u + 0x7fffu + ((u >> 16) & 1u);   // round-to-nearest-even
    return (unsigned short)(r >> 16);
}
__device__ __forceinline__ float bf2f(unsigned short u) {
    return __uint_as_float(((unsigned)u) << 16);
}
__device__ __forceinline__ float sigm(float x) {
    return 1.0f / (1.0f + __expf(-x));
}

// A-fragment: 16(M) x 32(K) bf16 from row-major [*, ld].
// Lanes 0-15 (row M=lane): K {k0..k0+7, k0+16..k0+23};
// lanes 16-31 (row M=lane-16): K {k0+8..+15, k0+24..+31}.
__device__ __forceinline__ bf16x16 load_a(const unsigned short* __restrict__ base,
                                          int ld, int row0, int k0, int lane) {
    int half = lane >> 4;
    int m    = lane & 15;
    const unsigned short* r = base + (size_t)(row0 + m) * ld;
    FragU f;
    f.q[0] = *(const uint4*)(r + k0 + half * 8);
    f.q[1] = *(const uint4*)(r + k0 + 16 + half * 8);
    return f.v;
}

// B-fragment: 32(K) x 16(N); B[k][n] = Wrow[n][k], W row-major [N_total, ld].
// Lanes 0-15: col N=lane, K {k0..k0+7, k0+16..k0+23}; lanes 16-31: same N, K {k0+8..+15, k0+24..+31}.
__device__ __forceinline__ bf16x16 load_b(const unsigned short* __restrict__ wbase,
                                          int ld, int col0, int k0, int lane) {
    int half = lane >> 4;
    int n    = lane & 15;
    const unsigned short* r = wbase + (size_t)(col0 + n) * ld + k0 + half * 8;
    FragU f;
    f.q[0] = *(const uint4*)(r);
    f.q[1] = *(const uint4*)(r + 16);
    return f.v;
}

__device__ __forceinline__ floatx8 wmma_bf(bf16x16 a, bf16x16 b, floatx8 c) {
    return __builtin_amdgcn_wmma_f32_16x16x32_bf16(
        false, a, false, b, (short)0, c, false, false);
}

// ---------------------------------------------------------------------------

// Vectorized fp32 -> bf16: 4 elements/thread. n4 = n/4 (all sizes %4 == 0).
__global__ void cvt_kernel(const float* __restrict__ src,
                           unsigned short* __restrict__ dst, int n4) {
    int i = blockIdx.x * blockDim.x + threadIdx.x;
    if (i < n4) {
        float4 f = ((const float4*)src)[i];
        uint2 p;
        p.x = (unsigned)f2bf(f.x) | ((unsigned)f2bf(f.y) << 16);
        p.y = (unsigned)f2bf(f.z) | ((unsigned)f2bf(f.w) << 16);
        ((uint2*)dst)[i] = p;
    }
}

// Input projections: M = B*T = 65536, K = I = 256, N = H = 512, gate = blockIdx.z.
// Block = 8 waves; wave w: 32(M) x 64(N) tile => 8 WMMA / K-step, register
// double-buffered over K.
__global__ void __launch_bounds__(256)
proj_kernel(const unsigned short* __restrict__ xbf,   // [B*T, I] bf16
            const unsigned short* __restrict__ Wc,    // [4, H, I] bf16
            const float* __restrict__ bz, const float* __restrict__ bs,
            const float* __restrict__ bff, const float* __restrict__ bo,
            unsigned short* __restrict__ proj)        // [4, T, B, H] bf16
{
    const int I = 256, H = 512, T = 512, BH = 128 * 512;
    int lane  = threadIdx.x & 31;
    int w     = threadIdx.x >> 5;
    int g     = blockIdx.z;
    int rowA  = (blockIdx.x * 8 + w) * 32;
    int col0  = blockIdx.y * 64;
    const unsigned short* wg = Wc + (size_t)g * H * I;

    floatx8 acc[2][4] = {};

    // Prime the pipeline.
    bf16x16 a0c = load_a(xbf, I, rowA,      0, lane);
    bf16x16 a1c = load_a(xbf, I, rowA + 16, 0, lane);
    bf16x16 bc[4];
#pragma unroll
    for (int nt = 0; nt < 4; ++nt) bc[nt] = load_b(wg, I, col0 + nt * 16, 0, lane);

#pragma unroll
    for (int k0 = 0; k0 < I; k0 += 32) {
        bf16x16 a0n = a0c, a1n = a1c, bn[4];
#pragma unroll
        for (int nt = 0; nt < 4; ++nt) bn[nt] = bc[nt];
        if (k0 + 32 < I) {            // compile-time under full unroll
            a0n = load_a(xbf, I, rowA,      k0 + 32, lane);
            a1n = load_a(xbf, I, rowA + 16, k0 + 32, lane);
#pragma unroll
            for (int nt = 0; nt < 4; ++nt)
                bn[nt] = load_b(wg, I, col0 + nt * 16, k0 + 32, lane);
        }
#pragma unroll
        for (int nt = 0; nt < 4; ++nt) acc[0][nt] = wmma_bf(a0c, bc[nt], acc[0][nt]);
#pragma unroll
        for (int nt = 0; nt < 4; ++nt) acc[1][nt] = wmma_bf(a1c, bc[nt], acc[1][nt]);
        a0c = a0n; a1c = a1n;
#pragma unroll
        for (int nt = 0; nt < 4; ++nt) bc[nt] = bn[nt];
    }

    const float* bias = (g == 0) ? bz : (g == 1) ? bs : (g == 2) ? bff : bo;
    int half = lane >> 4;
    int n    = lane & 15;
    unsigned short* pg = proj + (size_t)g * T * BH;
#pragma unroll
    for (int mt = 0; mt < 2; ++mt) {
#pragma unroll
        for (int nt = 0; nt < 4; ++nt) {
            int hcol = col0 + nt * 16 + n;
            float bv = bias[hcol];
#pragma unroll
            for (int v = 0; v < 8; ++v) {
                int m = rowA + mt * 16 + v + 8 * half;   // m = b*T + t
                int bidx = m >> 9;                       // / T
                int t    = m & 511;                      // % T
                pg[(size_t)t * BH + (size_t)bidx * H + hcol] = f2bf(acc[mt][nt][v] + bv);
            }
        }
    }
}

// One recurrence step. M = B = 128, K = N = H = 512, all 4 gates per wave
// (A-fragment shared). 128 waves: wid -> (mtile 0..7, 32-col strip 0..15).
__global__ void __launch_bounds__(128)
step_kernel(const unsigned short* __restrict__ proj,   // [4, T, B, H] bf16
            const unsigned short* __restrict__ Rc,     // [4, H, H] bf16
            const unsigned short* __restrict__ hprev,  // [B, H] bf16
            unsigned short* __restrict__ hnext,        // [B, H] bf16
            float* __restrict__ c,                     // [B, H]
            float* __restrict__ hacc,                  // [B, H]
            int t)
{
    const int H = 512, T = 512, BH = 128 * 512;
    int lane  = threadIdx.x & 31;
    int wid   = blockIdx.x * 4 + (threadIdx.x >> 5);   // 0..127
    int rowA  = (wid & 7) * 16;                        // batch tile
    int col0  = (wid >> 3) * 32;                       // hidden-col strip

    const unsigned short* pz = proj + ((size_t)0 * T + t) * BH;
    const unsigned short* ps = proj + ((size_t)1 * T + t) * BH;
    const unsigned short* pf = proj + ((size_t)2 * T + t) * BH;
    const unsigned short* po = proj + ((size_t)3 * T + t) * BH;

    // Prefetch epilogue operands (gfx1250 global_prefetch_b8) so they land in
    // cache while the 16-step GEMM pipeline runs. Each lane covers one row of
    // the 16x32 tile for this wave.
    {
        size_t ofs = (size_t)(rowA + (lane & 15)) * H + col0;
        __builtin_prefetch(pz + ofs, 0, 3);
        __builtin_prefetch(ps + ofs, 0, 3);
        __builtin_prefetch(pf + ofs, 0, 3);
        __builtin_prefetch(po + ofs, 0, 3);
        __builtin_prefetch(c + ofs, 1, 3);
        __builtin_prefetch(hacc + ofs, 1, 3);
    }

    const unsigned short* Rg[4] = {Rc,
                                   Rc + (size_t)1 * H * H,
                                   Rc + (size_t)2 * H * H,
                                   Rc + (size_t)3 * H * H};

    floatx8 acc[8] = {};                     // [gate*2 + ntile]

    bf16x16 ac = load_a(hprev, H, rowA, 0, lane);
    bf16x16 bc[8];
#pragma unroll
    for (int gq = 0; gq < 4; ++gq)
#pragma unroll
        for (int nt = 0; nt < 2; ++nt)
            bc[gq * 2 + nt] = load_b(Rg[gq], H, col0 + nt * 16, 0, lane);

#pragma unroll
    for (int k0 = 0; k0 < H; k0 += 32) {
        bf16x16 an = ac, bn[8];
#pragma unroll
        for (int j = 0; j < 8; ++j) bn[j] = bc[j];
        if (k0 + 32 < H) {                   // compile-time under full unroll
            an = load_a(hprev, H, rowA, k0 + 32, lane);
#pragma unroll
            for (int gq = 0; gq < 4; ++gq)
#pragma unroll
                for (int nt = 0; nt < 2; ++nt)
                    bn[gq * 2 + nt] = load_b(Rg[gq], H, col0 + nt * 16, k0 + 32, lane);
        }
#pragma unroll
        for (int j = 0; j < 8; ++j) acc[j] = wmma_bf(ac, bc[j], acc[j]);
        ac = an;
#pragma unroll
        for (int j = 0; j < 8; ++j) bc[j] = bn[j];
    }

    // Fused LSTM cell update + running-mean accumulation.
    int half = lane >> 4;
    int n    = lane & 15;
#pragma unroll
    for (int nt = 0; nt < 2; ++nt) {
        int hcol = col0 + nt * 16 + n;
#pragma unroll
        for (int v = 0; v < 8; ++v) {
            int bidx = rowA + v + 8 * half;
            size_t idx = (size_t)bidx * H + hcol;
            float z = tanhf(bf2f(pz[idx]) + acc[0 * 2 + nt][v]);
            float s = sigm (bf2f(ps[idx]) + acc[1 * 2 + nt][v]);
            float f = sigm (bf2f(pf[idx]) + acc[2 * 2 + nt][v]);
            float o = sigm (bf2f(po[idx]) + acc[3 * 2 + nt][v]);
            float cn = s * z + f * c[idx];
            c[idx] = cn;
            float hn = o * tanhf(cn);
            hacc[idx] += hn;
            hnext[idx] = f2bf(hn);
        }
    }
}

__global__ void final_kernel(const float* __restrict__ hacc,
                             float* __restrict__ out, int n, float scale) {
    int i = blockIdx.x * blockDim.x + threadIdx.x;
    if (i < n) out[i] = hacc[i] * scale;
}

// ---------------------------------------------------------------------------

extern "C" void kernel_launch(void* const* d_in, const int* in_sizes, int n_in,
                              void* d_out, int out_size, void* d_ws, size_t ws_size,
                              hipStream_t stream) {
    const int B = 128, T = 512, I = 256, H = 512;
    const int BH = B * H;

    const float* x  = (const float*)d_in[0];
    const float* W[4] = {(const float*)d_in[1], (const float*)d_in[2],
                         (const float*)d_in[3], (const float*)d_in[4]};
    const float* R[4] = {(const float*)d_in[5], (const float*)d_in[6],
                         (const float*)d_in[7], (const float*)d_in[8]};
    const float* bz = (const float*)d_in[9];
    const float* bs = (const float*)d_in[10];
    const float* bf = (const float*)d_in[11];
    const float* bo = (const float*)d_in[12];

    // Workspace layout (256B-aligned carves).
    uintptr_t p = ((uintptr_t)d_ws + 255) & ~(uintptr_t)255;
    auto carve = [&](size_t bytes) {
        uintptr_t r = p;
        p = (p + bytes + 255) & ~(uintptr_t)255;
        return (void*)r;
    };
    unsigned short* xbf  = (unsigned short*)carve((size_t)B * T * I * 2);   // 33.5 MB
    unsigned short* Wc   = (unsigned short*)carve((size_t)4 * H * I * 2);   // 1 MB
    unsigned short* Rc   = (unsigned short*)carve((size_t)4 * H * H * 2);   // 2 MB
    unsigned short* proj = (unsigned short*)carve((size_t)4 * T * BH * 2);  // 268 MB
    unsigned short* hb0  = (unsigned short*)carve((size_t)BH * 2);
    unsigned short* hb1  = (unsigned short*)carve((size_t)BH * 2);
    float*          cbuf = (float*)carve((size_t)BH * 4);
    float*          hacc = (float*)carve((size_t)BH * 4);

    // 1) fp32 -> bf16 staging (vectorized, 4 elems/thread).
    {
        int n4 = (B * T * I) / 4;
        cvt_kernel<<<(n4 + 255) / 256, 256, 0, stream>>>(x, xbf, n4);
        for (int g = 0; g < 4; ++g) {
            int nw4 = (H * I) / 4;
            cvt_kernel<<<(nw4 + 255) / 256, 256, 0, stream>>>(W[g], Wc + (size_t)g * H * I, nw4);
            int nr4 = (H * H) / 4;
            cvt_kernel<<<(nr4 + 255) / 256, 256, 0, stream>>>(R[g], Rc + (size_t)g * H * H, nr4);
        }
    }
    hipMemsetAsync(hb0,  0, (size_t)BH * 2, stream);
    hipMemsetAsync(cbuf, 0, (size_t)BH * 4, stream);
    hipMemsetAsync(hacc, 0, (size_t)BH * 4, stream);

    // 2) Input projections (compute-bound bf16 WMMA GEMM, 32x64 tile/wave).
    {
        dim3 grid((B * T) / (32 * 8), H / 64, 4);   // 256 x 8 x 4 blocks
        proj_kernel<<<grid, 256, 0, stream>>>(xbf, Wc, bz, bs, bf, bo, proj);
    }

    // 3) Sequential recurrence: one kernel per timestep (graph replay makes
    //    these launches cheap); ping-pong bf16 h buffers.
    for (int t = 0; t < T; ++t) {
        const unsigned short* hprev = (t & 1) ? hb1 : hb0;
        unsigned short*       hnext = (t & 1) ? hb0 : hb1;
        step_kernel<<<32, 128, 0, stream>>>(proj, Rc, hprev, hnext, cbuf, hacc, t);
    }

    // 4) h_bar = mean over T.
    final_kernel<<<(BH + 255) / 256, 256, 0, stream>>>(hacc, (float*)d_out, BH, 1.0f / T);
}
